// SwinTransformerBlock_3_branches_68375879352953
// MI455X (gfx1250) — compile-verified, matrix operand order checked by reading, not run on hardware
//
#include <hip/hip_runtime.h>
#include <cstdint>

// ---------------------------------------------------------------------------
// Swin block (3 branches) for MI455X / gfx1250, bf16 WMMA everywhere.
// GEMMs use double-buffered Tensor-Data-Mover tile loads (TENSORcnt pipelined).
// ---------------------------------------------------------------------------

typedef unsigned short u16;
typedef __attribute__((ext_vector_type(16))) __bf16 v16bf;
typedef __attribute__((ext_vector_type(8)))  float  v8f;

struct B32 { uint4 lo, hi; };          // 32 bytes == one v16bf fragment
struct U16x16 { u16 u[16]; };

static_assert(sizeof(v16bf) == 32, "v16bf must be 32 bytes");
static_assert(sizeof(B32) == 32, "B32 must be 32 bytes");

#define MTOK 100352                    // B * H * W tokens
#define DIMC 256
#define NHEAD 8
#define HD 32
#define NWIN 2048                      // B * 64 windows
#define ATT_SCALE 0.17677669529663687f // 32^-0.5

// ----------------------------- small helpers -------------------------------

static __device__ __forceinline__ u16 f2bf(float f) {
  unsigned u = __builtin_bit_cast(unsigned, f);
  unsigned r = u + 0x7FFFu + ((u >> 16) & 1u);   // round-to-nearest-even
  return (u16)(r >> 16);
}

static __device__ __forceinline__ v16bf frag_ld(const u16* p0, const u16* p1) {
  B32 t; t.lo = *(const uint4*)p0; t.hi = *(const uint4*)p1;
  return __builtin_bit_cast(v16bf, t);
}
static __device__ __forceinline__ v16bf frag_zero() {
  B32 t{}; return __builtin_bit_cast(v16bf, t);
}
static __device__ __forceinline__ v8f v8zero() {
  v8f z = {0.f,0.f,0.f,0.f,0.f,0.f,0.f,0.f}; return z;
}
static __device__ __forceinline__ v8f wmma_bf16(v16bf a, v16bf b, v8f c) {
  // D(16x16,f32) = A(16x32,bf16) x B(32x16,bf16) + C
  return __builtin_amdgcn_wmma_f32_16x16x32_bf16(false, a, false, b,
                                                 (short)0, c, false, false);
}

// token row for window w (0..2047), in-window position n (0..48)
static __device__ __forceinline__ int tok_row(int w, int n) {
  int b = w >> 6, wi = (w >> 3) & 7, wj = w & 7;
  int i = n / 7, j = n - i * 7;
  return b * 3136 + (wi * 7 + i) * 56 + (wj * 7 + j);
}

// ----------------------- Tensor Data Mover (gfx1250) -----------------------

#ifndef USE_TDM
#define USE_TDM 1
#endif
#if defined(__has_builtin)
# if __has_builtin(__builtin_amdgcn_tensor_load_to_lds) && \
     __has_builtin(__builtin_amdgcn_s_wait_tensorcnt)
#  define HAVE_TDM 1
# endif
#endif

#if USE_TDM && defined(HAVE_TDM)
typedef __attribute__((ext_vector_type(4))) unsigned int v4u;
typedef __attribute__((ext_vector_type(8))) int v8i;
typedef __attribute__((ext_vector_type(4))) int v4i;

// 2D tile load: 128 rows x 32 bf16 elements, row stride K elements,
// packed contiguously into LDS at byte offset ldsOff.
static __device__ __forceinline__ void tdm_load_tile(unsigned ldsOff,
                                                     const u16* g, int K) {
  unsigned long long ga = (unsigned long long)(uintptr_t)g;
  v4u g0;
  g0[0] = 1u;                                        // count=1, user mode
  g0[1] = ldsOff;                                    // lds_addr
  g0[2] = (unsigned)(ga & 0xFFFFFFFFull);            // global_addr[31:0]
  g0[3] = (unsigned)((ga >> 32) & 0x1FFFFFFull) | (2u << 30); // [56:32]|type=2
  v8i g1;
  g1[0] = 0x00010000;                                // data_size=1 (2B), no mcast
  g1[1] = (int)((unsigned)K << 16);                  // tensor_dim0 lo16
  g1[2] = (int)(((unsigned)K >> 16) | (128u << 16)); // dim0 hi16 | tensor_dim1 lo16
  g1[3] = (int)(32u << 16);                          // dim1 hi16=0 | tile_dim0=32
  g1[4] = 128;                                       // tile_dim1=128, tile_dim2=0
  g1[5] = K;                                         // tensor_dim0_stride lo32
  g1[6] = 0;                                         // stride hi16 | dim1_stride lo
  g1[7] = 0;
  v4i z4 = {0, 0, 0, 0};
  v8i z8 = {0, 0, 0, 0, 0, 0, 0, 0};
  // 6-arg form (this toolchain): (g0, g1, g2, g3, g4, cpol)
  __builtin_amdgcn_tensor_load_to_lds(g0, g1, z4, z4, z8, 0);
}
#endif

// ------------------------------- kernels -----------------------------------

__global__ void f32_to_bf16_kernel(const float* __restrict__ in,
                                   u16* __restrict__ out, int n) {
  int i = blockIdx.x * blockDim.x + threadIdx.x;
  if (i < n) out[i] = f2bf(in[i]);
}

// bias[h][n][m] = rpb_table[RPI[n][m]][h], RPI recomputed from window geometry
__global__ void rpb_gather_kernel(const float* __restrict__ rpb,
                                  float* __restrict__ biasg) {
  int idx = blockIdx.x * blockDim.x + threadIdx.x;
  if (idx >= NHEAD * 49 * 49) return;
  int h = idx / (49 * 49);
  int rem = idx % (49 * 49);
  int n = rem / 49, m = rem % 49;
  int i = n / 7, j = n % 7, i2 = m / 7, j2 = m % 7;
  int rpi = (i - i2 + 6) * 13 + (j - j2 + 6);
  biasg[idx] = rpb[rpi * NHEAD + h];
}

// LayerNorm over 256 features -> bf16. One wave per row, 8 rows per block.
__global__ __launch_bounds__(256)
void ln_bf16_kernel(const float* __restrict__ in, const float* __restrict__ g,
                    const float* __restrict__ b, u16* __restrict__ out) {
  const int lane = threadIdx.x & 31;
  const int row = blockIdx.x * 8 + (threadIdx.x >> 5);
  const float* p = in + (size_t)row * DIMC + lane * 8;
  float4 a = ((const float4*)p)[0];
  float4 c = ((const float4*)p)[1];
  float x[8] = {a.x, a.y, a.z, a.w, c.x, c.y, c.z, c.w};
  float s = 0.f, sq = 0.f;
#pragma unroll
  for (int e = 0; e < 8; ++e) { s += x[e]; sq += x[e] * x[e]; }
#pragma unroll
  for (int d = 1; d <= 16; d <<= 1) {
    s += __shfl_xor(s, d);
    sq += __shfl_xor(sq, d);
  }
  const float mean = s * (1.0f / 256.0f);
  const float var = sq * (1.0f / 256.0f) - mean * mean;
  const float rstd = rsqrtf(var + 1e-5f);
  u16* o = out + (size_t)row * DIMC + lane * 8;
#pragma unroll
  for (int e = 0; e < 8; ++e) {
    int col = lane * 8 + e;
    o[e] = f2bf((x[e] - mean) * rstd * g[col] + b[col]);
  }
}

// Generic C[M,N] = A[M,K](bf16) * W[N,K](bf16)^T + bias, with epilogues:
//   mode 0: bf16 out             (qkv)
//   mode 1: f32  out + residual  (proj / fc2)
//   mode 2: bf16 out, exact GELU (fc1)
// Block tile 128x128, 8 waves -> wave tile 32x64 (2x4 16x16 accumulators).
// Tiles are streamed by the Tensor Data Mover into double-buffered LDS:
// wave 0 issues the NEXT K-step's two descriptors, then s_wait_tensorcnt(2)
// so the two outstanding prefetches overlap this K-step's WMMAs.
__global__ __launch_bounds__(256)
void gemm_bf16_kernel(const u16* __restrict__ A, const u16* __restrict__ W,
                      const float* __restrict__ bias,
                      const float* __restrict__ resid,
                      void* __restrict__ outp, int M, int N, int K, int mode) {
  __shared__ u16 As[2][128][32];   // 16 KB
  __shared__ u16 Ws[2][128][32];   // 16 KB
  const int tid = threadIdx.x;
  const int lane = tid & 31, wv = tid >> 5;
  const int wvM = wv & 3, wvN = wv >> 2;       // 4 waves in M, 2 in N
  const int m0 = blockIdx.x * 128;
  const int n0 = blockIdx.y * 128;
  const int wm = wvM * 32, wn = wvN * 64;
  const int fr = lane & 15, kh = lane >> 4;
  const int nk = K / 32;

  v8f acc[2][4];
#pragma unroll
  for (int i = 0; i < 2; ++i)
#pragma unroll
    for (int j = 0; j < 4; ++j) acc[i][j] = v8zero();

#if USE_TDM && defined(HAVE_TDM)
  const unsigned ldsA0 = (unsigned)(uintptr_t)&As[0][0][0];
  const unsigned ldsA1 = (unsigned)(uintptr_t)&As[1][0][0];
  const unsigned ldsW0 = (unsigned)(uintptr_t)&Ws[0][0][0];
  const unsigned ldsW1 = (unsigned)(uintptr_t)&Ws[1][0][0];
  if (wv == 0) {                         // preload K-step 0 into buffer 0
    tdm_load_tile(ldsA0, A + (size_t)m0 * K, K);
    tdm_load_tile(ldsW0, W + (size_t)n0 * K, K);
  }
#endif

  for (int it = 0; it < nk; ++it) {
    const int cur = it & 1;
#if USE_TDM && defined(HAVE_TDM)
    // WAR guard: buffer cur^1 was last read in iteration it-1.
    __syncthreads();
    if (wv == 0) {
      if (it + 1 < nk) {                 // prefetch next K-step
        const int k1 = (it + 1) * 32;
        tdm_load_tile(cur ? ldsA0 : ldsA1, A + (size_t)m0 * K + k1, K);
        tdm_load_tile(cur ? ldsW0 : ldsW1, W + (size_t)n0 * K + k1, K);
        // in-order retirement: <=2 outstanding => current buffer is complete
        __builtin_amdgcn_s_wait_tensorcnt(2);
      } else {
        __builtin_amdgcn_s_wait_tensorcnt(0);
      }
    }
    __syncthreads();
#else
    __syncthreads();
    {
      const int k0 = it * 32;
      int r = tid >> 1, c = (tid & 1) * 16;   // 32 bytes per thread per tile
      const uint4* ga = (const uint4*)(A + (size_t)(m0 + r) * K + k0 + c);
      *(uint4*)&As[cur][r][c] = ga[0];
      *(uint4*)&As[cur][r][c + 8] = ga[1];
      const uint4* gw = (const uint4*)(W + (size_t)(n0 + r) * K + k0 + c);
      *(uint4*)&Ws[cur][r][c] = gw[0];
      *(uint4*)&Ws[cur][r][c + 8] = gw[1];
    }
    __syncthreads();
#endif
    v16bf af[2], wf[4];
#pragma unroll
    for (int i = 0; i < 2; ++i)
      af[i] = frag_ld(&As[cur][wm + i * 16 + fr][kh * 8],
                      &As[cur][wm + i * 16 + fr][16 + kh * 8]);
#pragma unroll
    for (int j = 0; j < 4; ++j)
      wf[j] = frag_ld(&Ws[cur][wn + j * 16 + fr][kh * 8],
                      &Ws[cur][wn + j * 16 + fr][16 + kh * 8]);
#pragma unroll
    for (int i = 0; i < 2; ++i)
#pragma unroll
      for (int j = 0; j < 4; ++j)
        acc[i][j] = wmma_bf16(af[i], wf[j], acc[i][j]);
  }

  const int rh = lane >> 4;
#pragma unroll
  for (int i = 0; i < 2; ++i)
#pragma unroll
    for (int j = 0; j < 4; ++j) {
      const int col = n0 + wn + j * 16 + fr;
      const float bv = bias[col];
#pragma unroll
      for (int r = 0; r < 8; ++r) {
        const int row = m0 + wm + i * 16 + r + 8 * rh;
        const size_t idx = (size_t)row * N + col;
        float v = acc[i][j][r] + bv;
        if (mode == 0) {
          ((u16*)outp)[idx] = f2bf(v);
        } else if (mode == 1) {
          ((float*)outp)[idx] = v + resid[idx];
        } else {
          v = 0.5f * v * (1.0f + erff(v * 0.70710678118654752f));
          ((u16*)outp)[idx] = f2bf(v);
        }
      }
    }
}

// ----------------------------- attention -----------------------------------

// A-fragment (rows) or B-fragment (cols) of Q/K straight from global qkv:
// the contiguous head-dim makes both 2x b128 loads per lane.
static __device__ __forceinline__ v16bf qk_frag(const u16* __restrict__ buf,
                                                int w, int h, int koff,
                                                int t16, int lane) {
  int fr = lane & 15, kh = lane >> 4;
  int n = t16 + fr;
  if (n >= 49) return frag_zero();
  const u16* p = buf + (size_t)tok_row(w, n) * 768 + koff + h * HD;
  return frag_ld(p + kh * 8, p + 16 + kh * 8);
}

// B-fragment of V (K = token dim, N = head dim): gathered u16 loads.
static __device__ __forceinline__ v16bf vfrag(const u16* __restrict__ buf,
                                              int w, int h, int nt, int ks,
                                              int lane) {
  int fr = lane & 15, kh = lane >> 4;
  int d = h * HD + nt * 16 + fr;
  U16x16 t;
#pragma unroll
  for (int e = 0; e < 16; ++e) {
    int k = ks * 32 + ((e < 8) ? (kh * 8 + e) : (16 + kh * 8 + (e - 8)));
    t.u[e] = (k < 49) ? buf[(size_t)tok_row(w, k) * 768 + 512 + d] : (u16)0;
  }
  return __builtin_bit_cast(v16bf, t);
}

// softmax over the 49 valid columns of S (held in C-layout accumulators),
// write P (bf16, zero-padded) into this head's LDS tile.
static __device__ __forceinline__ void softmax_store(v8f s[4][4],
                                                     const float* __restrict__ biasg,
                                                     int h, int lane,
                                                     u16* __restrict__ Ph) {
  const int fr = lane & 15, hp = lane >> 4;
#pragma unroll
  for (int mt = 0; mt < 4; ++mt)
#pragma unroll
    for (int r = 0; r < 8; ++r) {
      const int m = mt * 16 + r + 8 * hp;
      float x[4];
#pragma unroll
      for (int nt = 0; nt < 4; ++nt) {
        int n = nt * 16 + fr;
        if (n < 49) {
          float v = s[mt][nt][r] * ATT_SCALE;
          if (m < 49) v += biasg[(h * 49 + m) * 49 + n];
          x[nt] = v;
        } else {
          x[nt] = -1e30f;
        }
      }
      float mx = fmaxf(fmaxf(x[0], x[1]), fmaxf(x[2], x[3]));
#pragma unroll
      for (int d = 1; d <= 8; d <<= 1) mx = fmaxf(mx, __shfl_xor(mx, d));
      float sum = 0.f;
#pragma unroll
      for (int nt = 0; nt < 4; ++nt) { x[nt] = __expf(x[nt] - mx); sum += x[nt]; }
#pragma unroll
      for (int d = 1; d <= 8; d <<= 1) sum += __shfl_xor(sum, d);
      const float inv = 1.0f / sum;
#pragma unroll
      for (int nt = 0; nt < 4; ++nt) {
        int n = nt * 16 + fr;
        Ph[m * 64 + n] = f2bf((n < 49) ? x[nt] * inv : 0.f);
      }
    }
}

static __device__ __forceinline__ void load_aP(v16bf aP[4][2],
                                               const u16* __restrict__ Ph,
                                               int lane) {
  const int fr = lane & 15, kh = lane >> 4;
#pragma unroll
  for (int mt = 0; mt < 4; ++mt)
#pragma unroll
    for (int ks = 0; ks < 2; ++ks) {
      const u16* p = Ph + (mt * 16 + fr) * 64 + ks * 32;
      aP[mt][ks] = frag_ld(p + kh * 8, p + 16 + kh * 8);
    }
}

// O = P x V (49x64 * 64x32), store bf16 with window-reverse folded in.
static __device__ __forceinline__ void compute_store_o(
    const v16bf aP[4][2], const u16* __restrict__ vbuf, int w, int h, int lane,
    u16* __restrict__ out) {
  const int fr = lane & 15, hp = lane >> 4;
  v16bf bv[2][2];
#pragma unroll
  for (int nt = 0; nt < 2; ++nt)
#pragma unroll
    for (int ks = 0; ks < 2; ++ks) bv[nt][ks] = vfrag(vbuf, w, h, nt, ks, lane);
#pragma unroll
  for (int mt = 0; mt < 4; ++mt) {
    v8f oc[2] = {v8zero(), v8zero()};
#pragma unroll
    for (int ks = 0; ks < 2; ++ks)
#pragma unroll
      for (int nt = 0; nt < 2; ++nt)
        oc[nt] = wmma_bf16(aP[mt][ks], bv[nt][ks], oc[nt]);
#pragma unroll
    for (int nt = 0; nt < 2; ++nt)
#pragma unroll
      for (int r = 0; r < 8; ++r) {
        int m = mt * 16 + r + 8 * hp;
        if (m < 49)
          out[(size_t)tok_row(w, m) * DIMC + h * HD + nt * 16 + fr] =
              f2bf(oc[nt][r]);
      }
  }
}

// One block per window; one wave per head. Computes a2, a3 softmaxes and the
// three attention outputs o1=a2@v, o2=a2@v2, o3=a3@v2.
__global__ __launch_bounds__(256)
void attn_kernel(const u16* __restrict__ qkv, const u16* __restrict__ qkv2,
                 const float* __restrict__ biasg, u16* __restrict__ o1,
                 u16* __restrict__ o2, u16* __restrict__ o3) {
  __shared__ u16 P[NHEAD][64 * 64];   // 8 KB per head, 64 KB total
  const int w = blockIdx.x;
  const int h = threadIdx.x >> 5;
  const int lane = threadIdx.x & 31;
  u16* Ph = &P[h][0];

  // K2^T B-fragments shared by S2 and S3
  v16bf bk[4];
#pragma unroll
  for (int nt = 0; nt < 4; ++nt) bk[nt] = qk_frag(qkv2, w, h, 256, nt * 16, lane);

  v8f s[4][4];
#pragma unroll
  for (int i = 0; i < 4; ++i)
#pragma unroll
    for (int j = 0; j < 4; ++j) s[i][j] = v8zero();

  // S2 = Q2 . K2^T
#pragma unroll
  for (int mt = 0; mt < 4; ++mt) {
    v16bf a = qk_frag(qkv2, w, h, 0, mt * 16, lane);
#pragma unroll
    for (int nt = 0; nt < 4; ++nt) s[mt][nt] = wmma_bf16(a, bk[nt], s[mt][nt]);
  }
  softmax_store(s, biasg, h, lane, Ph);
  v16bf aP[4][2];
  load_aP(aP, Ph, lane);
  compute_store_o(aP, qkv, w, h, lane, o1);    // a2 @ v
  compute_store_o(aP, qkv2, w, h, lane, o2);   // a2 @ v2

  // S3 = Q . K2^T
#pragma unroll
  for (int i = 0; i < 4; ++i)
#pragma unroll
    for (int j = 0; j < 4; ++j) s[i][j] = v8zero();
#pragma unroll
  for (int mt = 0; mt < 4; ++mt) {
    v16bf a = qk_frag(qkv, w, h, 0, mt * 16, lane);
#pragma unroll
    for (int nt = 0; nt < 4; ++nt) s[mt][nt] = wmma_bf16(a, bk[nt], s[mt][nt]);
  }
  softmax_store(s, biasg, h, lane, Ph);
  load_aP(aP, Ph, lane);
  compute_store_o(aP, qkv2, w, h, lane, o3);   // a3 @ v2
}

// ------------------------------- launcher ----------------------------------

extern "C" void kernel_launch(void* const* d_in, const int* in_sizes, int n_in,
                              void* d_out, int out_size, void* d_ws,
                              size_t ws_size, hipStream_t stream) {
  const float* x      = (const float*)d_in[0];
  const float* x2     = (const float*)d_in[1];
  const float* n1g    = (const float*)d_in[2];
  const float* n1b    = (const float*)d_in[3];
  const float* qkv_w  = (const float*)d_in[4];
  const float* qkv_b  = (const float*)d_in[5];
  const float* rpb    = (const float*)d_in[6];
  const float* proj_w = (const float*)d_in[7];
  const float* proj_b = (const float*)d_in[8];
  const float* n2g    = (const float*)d_in[9];
  const float* n2b    = (const float*)d_in[10];
  const float* fc1_w  = (const float*)d_in[11];
  const float* fc1_b  = (const float*)d_in[12];
  const float* fc2_w  = (const float*)d_in[13];
  const float* fc2_b  = (const float*)d_in[14];

  char* ws = (char*)d_ws;
  const size_t SZ_QKV  = (size_t)MTOK * 768 * 2;   // 154,140,672
  const size_t SZ_ABF  = (size_t)MTOK * 256 * 2;   //  51,380,224
  const size_t SZ_AF32 = (size_t)MTOK * 256 * 4;   // 102,760,448
  const size_t OFF_QKV1 = 0;
  const size_t OFF_QKV2 = OFF_QKV1 + SZ_QKV;
  const size_t OFF_XN   = OFF_QKV2 + SZ_QKV;
  const size_t OFF_X2N  = OFF_XN + SZ_ABF;
  const size_t OFF_O1   = OFF_X2N + SZ_ABF;
  const size_t OFF_O2   = OFF_O1 + SZ_ABF;
  const size_t OFF_O3   = OFF_O2 + SZ_ABF;
  const size_t OFF_BIAS = OFF_O3 + SZ_ABF;
  const size_t OFF_W    = OFF_BIAS + (size_t)NHEAD * 49 * 49 * 4;
  // total ws footprint: OFF_W + 1,572,864 bytes  (~567 MB)

  u16* QKV1 = (u16*)(ws + OFF_QKV1);
  u16* QKV2 = (u16*)(ws + OFF_QKV2);
  u16* XN   = (u16*)(ws + OFF_XN);
  u16* X2N  = (u16*)(ws + OFF_X2N);
  u16* O1   = (u16*)(ws + OFF_O1);
  u16* O2   = (u16*)(ws + OFF_O2);
  u16* O3   = (u16*)(ws + OFF_O3);
  float* BIASG = (float*)(ws + OFF_BIAS);
  u16* WB = (u16*)(ws + OFF_W);
  u16* WQ = WB;                 // 768*256
  u16* WP = WB + 196608;        // 256*256
  u16* W1 = WB + 262144;        // 1024*256
  u16* W2 = WB + 524288;        // 256*1024
  // qkv buffers are dead after attention: r1..r3 alias them exactly.
  float* R1 = (float*)(ws + OFF_QKV1);
  float* R2 = (float*)(ws + OFF_QKV1 + SZ_AF32);
  float* R3 = (float*)(ws + OFF_QKV1 + 2 * SZ_AF32);
  u16* G  = XN;  // fc1 activation spans XN..O2 (4 * SZ_ABF == MTOK*1024*2)
  u16* HN = O3;  // LN(norm2) output

  float* out1 = (float*)d_out;
  float* out2 = out1 + (size_t)MTOK * DIMC;
  float* out3 = out2 + (size_t)MTOK * DIMC;

  dim3 blk(256);
  // weights -> bf16 (1.5 MB, negligible)
  f32_to_bf16_kernel<<<(196608 + 255) / 256, blk, 0, stream>>>(qkv_w, WQ, 196608);
  f32_to_bf16_kernel<<<(65536 + 255) / 256, blk, 0, stream>>>(proj_w, WP, 65536);
  f32_to_bf16_kernel<<<(262144 + 255) / 256, blk, 0, stream>>>(fc1_w, W1, 262144);
  f32_to_bf16_kernel<<<(262144 + 255) / 256, blk, 0, stream>>>(fc2_w, W2, 262144);
  rpb_gather_kernel<<<(NHEAD * 49 * 49 + 255) / 256, blk, 0, stream>>>(rpb, BIASG);

  // norm1
  ln_bf16_kernel<<<MTOK / 8, blk, 0, stream>>>(x, n1g, n1b, XN);
  ln_bf16_kernel<<<MTOK / 8, blk, 0, stream>>>(x2, n1g, n1b, X2N);

  // qkv projections
  dim3 gqkv(MTOK / 128, 768 / 128);
  gemm_bf16_kernel<<<gqkv, blk, 0, stream>>>(XN, WQ, qkv_b, nullptr, QKV1,
                                             MTOK, 768, 256, 0);
  gemm_bf16_kernel<<<gqkv, blk, 0, stream>>>(X2N, WQ, qkv_b, nullptr, QKV2,
                                             MTOK, 768, 256, 0);

  // windowed attention, 3 outputs
  attn_kernel<<<NWIN, blk, 0, stream>>>(QKV1, QKV2, BIASG, O1, O2, O3);

  // output projection + residual (fp32)
  dim3 g256(MTOK / 128, 256 / 128);
  gemm_bf16_kernel<<<g256, blk, 0, stream>>>(O1, WP, proj_b, x, R1,
                                             MTOK, 256, 256, 1);
  gemm_bf16_kernel<<<g256, blk, 0, stream>>>(O2, WP, proj_b, x2, R2,
                                             MTOK, 256, 256, 1);
  gemm_bf16_kernel<<<g256, blk, 0, stream>>>(O3, WP, proj_b, x, R3,
                                             MTOK, 256, 256, 1);

  // MLP (norm2 -> fc1+gelu -> fc2 + residual) per branch
  float* Rs[3] = {R1, R2, R3};
  float* outs[3] = {out1, out2, out3};
  dim3 gfc1(MTOK / 128, 1024 / 128);
  for (int i = 0; i < 3; ++i) {
    ln_bf16_kernel<<<MTOK / 8, blk, 0, stream>>>(Rs[i], n2g, n2b, HN);
    gemm_bf16_kernel<<<gfc1, blk, 0, stream>>>(HN, W1, fc1_b, nullptr, G,
                                               MTOK, 1024, 256, 2);
    gemm_bf16_kernel<<<g256, blk, 0, stream>>>(G, W2, fc2_b, Rs[i], outs[i],
                                               MTOK, 256, 1024, 1);
  }
  (void)in_sizes; (void)n_in; (void)out_size; (void)ws_size;
}
